// ScaledDotProductAttention_7490422964284
// MI455X (gfx1250) — matrix operand order repaired and checked by reading, not compiled, not run on hardware
//
#include <hip/hip_runtime.h>

typedef __attribute__((ext_vector_type(16))) _Float16 v16h;
typedef __attribute__((ext_vector_type(8)))  float    v8f;
typedef __attribute__((ext_vector_type(4)))  float    v4f;

static constexpr int NB = 4;     // batch
static constexpr int NH = 16;    // heads
static constexpr int NS = 2048;  // sequence
static constexpr int ND = 128;   // head dim
static constexpr float SCALE = 0.08838834764831845f; // 1/sqrt(128)

union HFrag {
    _Float16 h[16];
    unsigned u[8];
    v16h     v;
};

// convert 8 consecutive fp32 (32B-aligned) to 8 f16 fragment slots
__device__ __forceinline__ void cvt8(const float* __restrict__ p, _Float16* h) {
    v4f x = ((const v4f*)p)[0];
    v4f y = ((const v4f*)p)[1];
    h[0] = (_Float16)x[0]; h[1] = (_Float16)x[1];
    h[2] = (_Float16)x[2]; h[3] = (_Float16)x[3];
    h[4] = (_Float16)y[0]; h[5] = (_Float16)y[1];
    h[6] = (_Float16)y[2]; h[7] = (_Float16)y[3];
}

__global__ __launch_bounds__(256)
void sdpa_wmma_kernel(const float* __restrict__ Q, const float* __restrict__ K,
                      const float* __restrict__ V, const int* __restrict__ Msk,
                      float* __restrict__ Out, float* __restrict__ Attn)
{
    __shared__ float    red[8][16];            // cross-wave softmax partials
    __shared__ float    outAcc[16][ND];        // 8 KB cross-wave PV accumulator
    __shared__ _Float16 stage[8][16][34];      // per-wave 16x32 prob tile, padded

    const int tid  = threadIdx.x;
    const int wid  = tid >> 5;        // wave 0..7
    const int lane = tid & 31;
    const int l16  = lane & 15;
    const int half = lane >> 4;       // 0: lanes 0-15, 1: lanes 16-31

    const int q0 = blockIdx.x * 16;   // query-row tile
    const int h  = blockIdx.y;
    const int b  = blockIdx.z;
    const size_t bh = (size_t)b * NH + h;

    const float* Qb = Q + bh * (size_t)NS * ND;
    const float* Kb = K + bh * (size_t)NS * ND;
    const float* Vb = V + bh * (size_t)NS * ND;
    const int*   Mb = Msk + (size_t)b * NS * NS;
    float*       Ob = Out  + bh * (size_t)NS * ND;
    float*       Ab = Attn + bh * (size_t)NS * NS;

    const int colBase = wid * 256;    // this wave's key-column slab

    // zero cross-wave output accumulator (covered by softmax barrier below)
    for (int i = tid; i < 16 * ND; i += 256) ((float*)outAcc)[i] = 0.0f;

    // ---------------- Q A-fragments (held for the whole QK^T phase) ----------
    // A layout (16-bit, 16x32): lane -> row M = l16; K runs: [half*8,+8) and [16+half*8,+8)
    HFrag aq[4];
    {
        const float* qrow = Qb + (size_t)(q0 + l16) * ND;
        #pragma unroll
        for (int kc = 0; kc < 4; ++kc) {
            cvt8(qrow + kc * 32 + half * 8,      &aq[kc].h[0]);
            cvt8(qrow + kc * 32 + 16 + half * 8, &aq[kc].h[8]);
        }
    }

    // ---------------- scores = Q K^T for 16 column tiles ---------------------
    v8f acc[16];
    #pragma unroll
    for (int nt = 0; nt < 16; ++nt) {
        v8f c = {};
        #pragma unroll
        for (int kc = 0; kc < 4; ++kc) {
            // B layout (32x16): lane -> col N = l16 (key), K runs: kc*32+[half*8,+8) and kc*32+[16+half*8,+8) (head dim)
            HFrag bk;
            const float* krow = Kb + (size_t)(colBase + nt * 16 + l16) * ND + kc * 32 + half * 8;
            cvt8(krow,     &bk.h[0]);
            cvt8(krow + 16, &bk.h[8]);
            c = __builtin_amdgcn_wmma_f32_16x16x32_f16(false, aq[kc].v, false, bk.v,
                                                       (short)0, c, false, false);
        }
        acc[nt] = c;
    }

    // ---------------- scale + mask + row max ---------------------------------
    // C layout: lane holds col N = l16 (+tile), rows M = r + half*8, r = 0..7
    float gmax[8];
    #pragma unroll
    for (int r = 0; r < 8; ++r) gmax[r] = -3.0e38f;
    #pragma unroll
    for (int nt = 0; nt < 16; ++nt) {
        const int col = colBase + nt * 16 + l16;
        #pragma unroll
        for (int r = 0; r < 8; ++r) {
            const int row = r + half * 8;
            const int m = Mb[(size_t)(q0 + row) * NS + col];
            float s = (m != 0) ? acc[nt][r] * SCALE : -1.0e9f;
            acc[nt][r] = s;
            gmax[r] = fmaxf(gmax[r], s);
        }
    }
    #pragma unroll
    for (int r = 0; r < 8; ++r) {
        #pragma unroll
        for (int off = 1; off < 16; off <<= 1)
            gmax[r] = fmaxf(gmax[r], __shfl_xor(gmax[r], off, 32));
    }
    if (l16 == 0) {
        #pragma unroll
        for (int r = 0; r < 8; ++r) red[wid][half * 8 + r] = gmax[r];
    }
    __syncthreads();
    #pragma unroll
    for (int r = 0; r < 8; ++r) {
        float m = red[0][half * 8 + r];
        #pragma unroll
        for (int w = 1; w < 8; ++w) m = fmaxf(m, red[w][half * 8 + r]);
        gmax[r] = m;
    }
    __syncthreads();   // red about to be reused for sums

    // ---------------- exp + row sum ------------------------------------------
    float gsum[8];
    #pragma unroll
    for (int r = 0; r < 8; ++r) gsum[r] = 0.0f;
    #pragma unroll
    for (int nt = 0; nt < 16; ++nt) {
        #pragma unroll
        for (int r = 0; r < 8; ++r) {
            float p = __expf(acc[nt][r] - gmax[r]);
            acc[nt][r] = p;
            gsum[r] += p;
        }
    }
    #pragma unroll
    for (int r = 0; r < 8; ++r) {
        #pragma unroll
        for (int off = 1; off < 16; off <<= 1)
            gsum[r] += __shfl_xor(gsum[r], off, 32);
    }
    if (l16 == 0) {
        #pragma unroll
        for (int r = 0; r < 8; ++r) red[wid][half * 8 + r] = gsum[r];
    }
    __syncthreads();
    #pragma unroll
    for (int r = 0; r < 8; ++r) {
        float s = red[0][half * 8 + r];
        #pragma unroll
        for (int w = 1; w < 8; ++w) s += red[w][half * 8 + r];
        gsum[r] = 1.0f / s;   // rinv
    }

    // ---------------- normalize, write full attention matrix -----------------
    #pragma unroll
    for (int nt = 0; nt < 16; ++nt) {
        const int col = colBase + nt * 16 + l16;
        #pragma unroll
        for (int r = 0; r < 8; ++r) {
            float p = acc[nt][r] * gsum[r];
            acc[nt][r] = p;
            Ab[(size_t)(q0 + r + half * 8) * NS + col] = p;
        }
    }

    // ---------------- out partial = P(16 x 256-slab) @ V ---------------------
    v8f oacc[8];
    #pragma unroll
    for (int d = 0; d < 8; ++d) oacc[d] = (v8f){};

    _Float16 (*st)[34] = stage[wid];
    #pragma unroll
    for (int c = 0; c < 8; ++c) {
        // stage probs for key chunk [colBase+c*32, +32) as f16 row-major in LDS
        #pragma unroll
        for (int tt = 0; tt < 2; ++tt) {
            const int nt = 2 * c + tt;
            #pragma unroll
            for (int r = 0; r < 8; ++r)
                st[r + half * 8][tt * 16 + l16] = (_Float16)acc[nt][r];
        }
        // A fragment (rows = queries, K = 32 keys) from own staging tile
        // (same-wave LDS store->load: hardware keeps DS ops in order)
        HFrag pa;
        {
            const unsigned* sp0 = (const unsigned*)&st[l16][half * 16];
            const unsigned* sp1 = (const unsigned*)&st[l16][half * 16 + 8];
            #pragma unroll
            for (int j = 0; j < 4; ++j) { pa.u[j] = sp0[j]; pa.u[4 + j] = sp1[j]; }
        }
        // B fragments from V: lane -> col N = head dim, K = key rows (strided rows,
        // coalesced across lanes: 16 consecutive d per key row)
        const int kb = colBase + c * 32 + half * 16;
        #pragma unroll
        for (int d = 0; d < 8; ++d) {
            HFrag vb;
            const float* vp = Vb + (size_t)kb * ND + d * 16 + l16;
            #pragma unroll
            for (int j = 0; j < 8; ++j) {
                vb.h[2 * j]     = (_Float16)vp[(size_t)(2 * j) * ND];
                vb.h[2 * j + 1] = (_Float16)vp[(size_t)(2 * j + 1) * ND];
            }
            oacc[d] = __builtin_amdgcn_wmma_f32_16x16x32_f16(false, pa.v, false, vb.v,
                                                             (short)0, oacc[d], false, false);
        }
    }

    // ---------------- cross-wave reduce + store out --------------------------
    #pragma unroll
    for (int d = 0; d < 8; ++d) {
        #pragma unroll
        for (int r = 0; r < 8; ++r)
            atomicAdd(&outAcc[r + half * 8][d * 16 + l16], oacc[d][r]);
    }
    __syncthreads();
    for (int i = tid; i < 16 * ND; i += 256) {
        const int row = i >> 7;
        const int dd  = i & (ND - 1);
        Ob[(size_t)(q0 + row) * ND + dd] = outAcc[row][dd];
    }
}

extern "C" void kernel_launch(void* const* d_in, const int* in_sizes, int n_in,
                              void* d_out, int out_size, void* d_ws, size_t ws_size,
                              hipStream_t stream) {
    (void)in_sizes; (void)n_in; (void)out_size; (void)d_ws; (void)ws_size;
    const float* Q   = (const float*)d_in[0];
    const float* K   = (const float*)d_in[1];
    const float* V   = (const float*)d_in[2];
    const int*   Msk = (const int*)d_in[3];
    // d_in[4] = softmax_dim (-1), fixed by construction

    float* Out  = (float*)d_out;
    float* Attn = Out + (size_t)NB * NH * NS * ND;   // outputs concatenated: out, attention

    dim3 grid(NS / 16, NH, NB);   // 128 x 16 x 4 workgroups
    dim3 block(256);              // 8 wave32 waves
    hipLaunchKernelGGL(sdpa_wmma_kernel, grid, block, 0, stream, Q, K, V, Msk, Out, Attn);
}
